// HypRelEncoder_44667659878559
// MI455X (gfx1250) — compile-verified, matrix operand order checked by reading, not run on hardware
//
#include <hip/hip_runtime.h>
#include <cstddef>

#define DDIM 200
#define KPAD 224
#define KT_TILES 7      // K tiles of 32 (224 = 7*32)
#define NT_TILES 13     // N tiles of 16 (208 = 13*16)
#define N_ENT 200000
#define EDIR 500000
#define NQDIR 400000
#define BATCH 4096
#define QPAIR 6
#define FRAG_SZ 512                         // bf16 elems per 16x32 packed B tile
#define PACK_W_ELEMS (KT_TILES * NT_TILES * FRAG_SZ)  // 46592 ushorts per weight

typedef __attribute__((ext_vector_type(16))) __bf16 v16bf;
typedef __attribute__((ext_vector_type(8)))  float  v8f;

// ---------------- helpers ----------------

__device__ __forceinline__ unsigned short f2bf(float f) {
  union { float f; unsigned int u; } c; c.f = f;
  unsigned int r = c.u + 0x7FFFu + ((c.u >> 16) & 1u);   // round-to-nearest-even
  return (unsigned short)(r >> 16);
}

// A fragment (16x32 bf16) from LDS row-major staging (pitch KPAD bf16).
// lane = nl + 16*h holds row M=nl; VGPR p, elem e -> K = kb(p) + 8h + e.
__device__ __forceinline__ v16bf load_a_frag(const unsigned short* rowbase, int kt, int h) {
  union { v16bf v; unsigned int u[8]; } fr;
  const unsigned short* t = rowbase + kt * 32 + 8 * h;
#pragma unroll
  for (int p = 0; p < 8; ++p) {
    int kb = (p < 4) ? (2 * p) : (16 + 2 * (p - 4));
    fr.u[p] = *(const unsigned int*)(t + kb);            // 2 packed bf16, dword aligned
  }
  return fr.v;
}

// B fragment: pre-packed so each lane reads 16 contiguous bf16 (32B).
__device__ __forceinline__ v16bf load_b_frag(const unsigned short* packed, int kt, int nt, int lane) {
  union { v16bf v; uint4 q[2]; } fr;
  const uint4* g = (const uint4*)(packed + (size_t)(kt * NT_TILES + nt) * FRAG_SZ + lane * 16);
  fr.q[0] = g[0]; fr.q[1] = g[1];
  return fr.v;
}

__device__ __forceinline__ v8f wmma_bf16(v16bf a, v16bf b, v8f c) {
  return __builtin_amdgcn_wmma_f32_16x16x32_bf16(false, a, false, b, (short)0, c, false, false);
}

__device__ __forceinline__ unsigned lds_offset(const void* p) {
  return (unsigned)(size_t)p;   // LDS aperture: offset lives in addr[31:0]
}

// ---------------- small utility kernels ----------------

__global__ void zero_kernel(float* __restrict__ p, long n) {
  long g = (long)blockIdx.x * blockDim.x + threadIdx.x;
  if (g < n) p[g] = 0.f;
}

// rel_all = concat(r[200,200], loop_rel[1,200])
__global__ void build_rel_all_kernel(const float* __restrict__ r, const float* __restrict__ loop_rel,
                                     float* __restrict__ rel_all) {
  int g = blockIdx.x * blockDim.x + threadIdx.x;
  if (g < 201 * DDIM) rel_all[g] = (g < 200 * DDIM) ? r[g] : loop_rel[g - 200 * DDIM];
}

// pack one 200x200 fp32 weight into bf16 WMMA-B fragments (zero padded to 224x208)
__global__ void pack_w_kernel(const float* __restrict__ W, unsigned short* __restrict__ out) {
  int tile = blockIdx.x;
  int kt = tile / NT_TILES, nt = tile % NT_TILES;
  int l = threadIdx.x, nl = l & 15, h = l >> 4;
  unsigned short* o = out + (size_t)tile * FRAG_SZ + l * 16;
  int N = nt * 16 + nl;
#pragma unroll
  for (int p = 0; p < 8; ++p) {
    int kb = (p < 4) ? (2 * p) : (16 + 2 * (p - 4));
#pragma unroll
    for (int e = 0; e < 2; ++e) {
      int K = kt * 32 + kb + 8 * h + e;
      float f = (K < DDIM && N < DDIM) ? W[K * DDIM + N] : 0.f;
      o[2 * p + e] = f2bf(f);
    }
  }
}

__global__ void deg_count_kernel(const int* __restrict__ src, const int* __restrict__ dst,
                                 float* __restrict__ dega, float* __restrict__ degb, int E) {
  int g = blockIdx.x * blockDim.x + threadIdx.x;
  if (g < E) {
    atomicAdd(dega + src[g], 1.f);
    atomicAdd(degb + dst[g], 1.f);
  }
}

__global__ void edge_norm_kernel(const int* __restrict__ src, const int* __restrict__ dst,
                                 const float* __restrict__ dega, const float* __restrict__ degb,
                                 float* __restrict__ nin, float* __restrict__ nout, int E) {
  int g = blockIdx.x * blockDim.x + threadIdx.x;
  if (g < E) {
    int s = src[g], d = dst[g];
    float as = dega[s], ad = dega[d];
    float bs = degb[s], bd = degb[d];
    float ias = (as > 0.f) ? rsqrtf(fmaxf(as, 1.f)) : 0.f;
    float iad = (ad > 0.f) ? rsqrtf(fmaxf(ad, 1.f)) : 0.f;
    float ibs = (bs > 0.f) ? rsqrtf(fmaxf(bs, 1.f)) : 0.f;
    float ibd = (bd > 0.f) ? rsqrtf(fmaxf(bd, 1.f)) : 0.f;
    nin[g]  = ias * iad;   // dir-in: deg over src
    nout[g] = ibd * ibs;   // dir-out: src'=dst, deg over dst
  }
}

// q_agg(bf16)[eid] += bf16(x[q_e] * rel_all[q_r]) via packed bf16 atomics.
// One block per qualifier; lane c handles columns 2c, 2c+1 (100 active lanes).
__global__ void qual_scatter_kernel(const int* __restrict__ qr, const int* __restrict__ qe,
                                    const int* __restrict__ qid,
                                    const float* __restrict__ x, const float* __restrict__ rel_all,
                                    unsigned short* __restrict__ q_agg, int NQ) {
  int j = blockIdx.x, c = threadIdx.x;
  if (j < NQ && c < DDIM / 2) {
    int e = qe[j], r = qid ? qr[j] : 0, id = qid[j];
    int cc = 2 * c;
    const float* xp = x + (size_t)e * DDIM + cc;
    const float* rp = rel_all + (size_t)r * DDIM + cc;
    unsigned pk = (unsigned)f2bf(xp[0] * rp[0]) | ((unsigned)f2bf(xp[1] * rp[1]) << 16);
    unsigned short* dstp = q_agg + (size_t)id * DDIM + cc;
    asm volatile("global_atomic_pk_add_bf16 %0, %1, off"
                 :: "v"(dstp), "v"(pk) : "memory");
  }
}

// ---------------- fused per-edge-tile double GEMM (WMMA) ----------------
// For 16 edges: rel_e = 0.8*rel_all[et] + 0.2*(q_agg @ w_q);
//               msg   = (x[src] * rel_e) @ w;  node_acc[dst] += norm * msg
__global__ void edge_msg_kernel(const int* __restrict__ S, const int* __restrict__ Dv,
                                const int* __restrict__ ET, const float* __restrict__ normE,
                                const unsigned short* __restrict__ q_agg, const float* __restrict__ x,
                                const float* __restrict__ rel_all,
                                const unsigned short* __restrict__ wq_p,
                                const unsigned short* __restrict__ w_p,
                                float* __restrict__ node_acc, int E) {
  __shared__ int   s_src[16], s_dst[16], s_et[16];
  __shared__ float s_nrm[16];
  __shared__ __align__(16) unsigned short A[16 * KPAD];
  __shared__ __align__(16) unsigned short T[16 * KPAD];

  int l = threadIdx.x;
  int e0 = blockIdx.x * 16;

  // --- async DMA: 16 contiguous bf16 rows (6400B) of q_agg -> padded LDS ---
  {
    const unsigned short* gsrc = q_agg + (size_t)e0 * DDIM;  // rows are contiguous
    unsigned abase = lds_offset(&A[0]);
#pragma unroll
    for (int i = 0; i < 13; ++i) {
      int b = i * 512 + l * 16;                // byte offset within the 6400B block
      if (b < 16 * DDIM * 2) {
        int row = b / (DDIM * 2), col = b % (DDIM * 2);
        unsigned lds = abase + row * (KPAD * 2) + col;
        unsigned long ga = (unsigned long)gsrc + (unsigned)b;
        asm volatile("global_load_async_to_lds_b128 %0, %1, off"
                     :: "v"(lds), "v"(ga) : "memory");
      }
    }
  }

  if (l < 16) {
    int e = e0 + l;
    bool ok = e < E;
    s_src[l] = ok ? S[e]  : 0;
    s_dst[l] = ok ? Dv[e] : 0;
    s_et[l]  = ok ? ET[e] : 0;
    s_nrm[l] = ok ? normE[e] : 0.f;
  }
  // zero A's K-padding (cols 200..223, 12 dwords/row) and all of T
  for (int p = l; p < 16 * 12; p += 32) {
    int row = p / 12, cu = p % 12;
    *(unsigned int*)&A[row * KPAD + DDIM + 2 * cu] = 0u;
  }
  for (int idx = l; idx < 16 * KPAD / 2; idx += 32)
    *(unsigned int*)&T[2 * idx] = 0u;

  asm volatile("s_wait_asynccnt 0x0" ::: "memory");
  __syncthreads();

  int nl = l & 15, h = l >> 4;
  const unsigned short* Arow = A + nl * KPAD;

  // stage 1: T = bf16( x[src] * (0.8*rel + 0.2*(A @ wq)) )
  for (int nt = 0; nt < NT_TILES; ++nt) {
    v8f acc = {0.f, 0.f, 0.f, 0.f, 0.f, 0.f, 0.f, 0.f};
#pragma unroll
    for (int kt = 0; kt < KT_TILES; ++kt)
      acc = wmma_bf16(load_a_frag(Arow, kt, h), load_b_frag(wq_p, kt, nt, l), acc);
#pragma unroll
    for (int r = 0; r < 8; ++r) {                 // C layout: lane n=nl, row m=r+8h
      int m = r + 8 * h;
      int n = nt * 16 + nl;
      float t = 0.f;
      if (n < DDIM && (e0 + m) < E) {
        float xv = x[(size_t)s_src[m] * DDIM + n];
        float rv = rel_all[(size_t)s_et[m] * DDIM + n];
        t = xv * (0.8f * rv + 0.2f * acc[r]);
      }
      T[m * KPAD + nt * 16 + nl] = f2bf(t);
    }
  }
  __syncthreads();

  const unsigned short* Trow = T + nl * KPAD;
  // stage 2: node_acc[dst] += norm * (T @ w)
  for (int nt = 0; nt < NT_TILES; ++nt) {
    v8f acc = {0.f, 0.f, 0.f, 0.f, 0.f, 0.f, 0.f, 0.f};
#pragma unroll
    for (int kt = 0; kt < KT_TILES; ++kt)
      acc = wmma_bf16(load_a_frag(Trow, kt, h), load_b_frag(w_p, kt, nt, l), acc);
#pragma unroll
    for (int r = 0; r < 8; ++r) {
      int m = r + 8 * h;
      int n = nt * 16 + nl;
      if (n < DDIM && (e0 + m) < E)
        atomicAdd(node_acc + (size_t)s_dst[m] * DDIM + n, s_nrm[m] * acc[r]);
    }
  }
}

// ---------------- loop GEMM + combine + tanh ----------------
__global__ void loop_final_kernel(const float* __restrict__ x_in, const float* __restrict__ loop_rel,
                                  const unsigned short* __restrict__ wl_p,
                                  const float* __restrict__ node_acc, const float* __restrict__ bias,
                                  float* __restrict__ x_out, int N) {
  __shared__ __align__(16) unsigned short A[16 * KPAD];
  int l = threadIdx.x;
  int v0 = blockIdx.x * 16;
  for (int idx = l; idx < 16 * KPAD; idx += 32) {
    int m = idx / KPAD, c = idx % KPAD;
    int v = v0 + m;
    float f = (v < N && c < DDIM) ? x_in[(size_t)v * DDIM + c] * loop_rel[c] : 0.f;
    A[idx] = f2bf(f);
  }
  __syncthreads();
  int nl = l & 15, h = l >> 4;
  const unsigned short* Arow = A + nl * KPAD;
  for (int nt = 0; nt < NT_TILES; ++nt) {
    v8f acc = {0.f, 0.f, 0.f, 0.f, 0.f, 0.f, 0.f, 0.f};
#pragma unroll
    for (int kt = 0; kt < KT_TILES; ++kt)
      acc = wmma_bf16(load_a_frag(Arow, kt, h), load_b_frag(wl_p, kt, nt, l), acc);
#pragma unroll
    for (int r = 0; r < 8; ++r) {
      int m = r + 8 * h, n = nt * 16 + nl, v = v0 + m;
      if (v < N && n < DDIM) {
        float o = (node_acc[(size_t)v * DDIM + n] + acc[r]) * (1.f / 3.f) + bias[n];
        x_out[(size_t)v * DDIM + n] = tanhf(o);
      }
    }
  }
}

// ---------------- rel GEMM: r_out = rel_all[:200] @ w_rel ----------------
__global__ void rel_gemm_kernel(const float* __restrict__ rel_all,
                                const unsigned short* __restrict__ wr_p,
                                float* __restrict__ r_out) {
  __shared__ __align__(16) unsigned short A[16 * KPAD];
  int l = threadIdx.x;
  int v0 = blockIdx.x * 16;
  for (int idx = l; idx < 16 * KPAD; idx += 32) {
    int m = idx / KPAD, c = idx % KPAD;
    int v = v0 + m;
    float f = (v < 200 && c < DDIM) ? rel_all[(size_t)v * DDIM + c] : 0.f;
    A[idx] = f2bf(f);
  }
  __syncthreads();
  int nl = l & 15, h = l >> 4;
  const unsigned short* Arow = A + nl * KPAD;
  for (int nt = 0; nt < NT_TILES; ++nt) {
    v8f acc = {0.f, 0.f, 0.f, 0.f, 0.f, 0.f, 0.f, 0.f};
#pragma unroll
    for (int kt = 0; kt < KT_TILES; ++kt)
      acc = wmma_bf16(load_a_frag(Arow, kt, h), load_b_frag(wr_p, kt, nt, l), acc);
#pragma unroll
    for (int r = 0; r < 8; ++r) {
      int m = r + 8 * h, n = nt * 16 + nl, v = v0 + m;
      if (v < 200 && n < DDIM) r_out[(size_t)v * DDIM + n] = acc[r];
    }
  }
}

// ---------------- output gathers ----------------
__global__ void gather_kernel(const float* __restrict__ tab, const int* __restrict__ ix,
                              int stride, int off, float* __restrict__ out, int nrows) {
  int g = blockIdx.x * blockDim.x + threadIdx.x;
  int total = nrows * DDIM;
  if (g < total) {
    int i = g / DDIM, c = g % DDIM;
    out[g] = tab[(size_t)ix[i * stride + off] * DDIM + c];
  }
}

// ---------------- host orchestration ----------------

static inline unsigned cdivu(long a, long b) { return (unsigned)((a + b - 1) / b); }
static inline size_t alignup(size_t x) { return (x + 255) & ~(size_t)255; }

static void run_layer(hipStream_t stream,
                      const float* x_in, const float* r_in,
                      const float* loop_rel, const float* bias,
                      const unsigned short* p_win, const unsigned short* p_wout,
                      const unsigned short* p_wloop, const unsigned short* p_wrel,
                      const unsigned short* p_wq,
                      const int* eidx, const int* etyp, const int* quals,
                      const float* nin, const float* nout,
                      unsigned short* q_agg, float* node_acc, float* rel_all,
                      float* x_out, float* r_out) {
  const int E = EDIR, NQ = NQDIR;
  const long QAGG_F32 = (long)E * DDIM / 2;   // bf16 buffer zeroed as floats
  build_rel_all_kernel<<<cdivu(201 * DDIM, 256), 256, 0, stream>>>(r_in, loop_rel, rel_all);
  zero_kernel<<<cdivu((long)N_ENT * DDIM, 256), 256, 0, stream>>>(node_acc, (long)N_ENT * DDIM);

  const int* S_in  = eidx;            const int* D_in  = eidx + 2 * E;
  const int* S_out = eidx + E;        const int* D_out = eidx + 3 * E;
  const int* qr  = quals;
  const int* qe  = quals + 2 * NQ;
  const int* qid = quals + 4 * NQ;

  // in direction
  zero_kernel<<<cdivu(QAGG_F32, 256), 256, 0, stream>>>((float*)q_agg, QAGG_F32);
  qual_scatter_kernel<<<NQ, 128, 0, stream>>>(qr, qe, qid, x_in, rel_all, q_agg, NQ);
  edge_msg_kernel<<<E / 16, 32, 0, stream>>>(S_in, D_in, etyp, nin, q_agg, x_in, rel_all,
                                             p_wq, p_win, node_acc, E);
  // out direction
  zero_kernel<<<cdivu(QAGG_F32, 256), 256, 0, stream>>>((float*)q_agg, QAGG_F32);
  qual_scatter_kernel<<<NQ, 128, 0, stream>>>(qr + NQ, qe + NQ, qid + NQ, x_in, rel_all, q_agg, NQ);
  edge_msg_kernel<<<E / 16, 32, 0, stream>>>(S_out, D_out, etyp + E, nout, q_agg, x_in, rel_all,
                                             p_wq, p_wout, node_acc, E);

  loop_final_kernel<<<N_ENT / 16, 32, 0, stream>>>(x_in, loop_rel, p_wloop, node_acc, bias, x_out, N_ENT);
  rel_gemm_kernel<<<13, 32, 0, stream>>>(rel_all, p_wrel, r_out);
}

extern "C" void kernel_launch(void* const* d_in, const int* in_sizes, int n_in,
                              void* d_out, int out_size, void* d_ws, size_t ws_size,
                              hipStream_t stream) {
  const float* ent  = (const float*)d_in[0];
  const float* rel  = (const float*)d_in[1];
  const float* W[10];
  for (int i = 0; i < 10; ++i) W[i] = (const float*)d_in[2 + i];  // in1,out1,loop1,rel1,q1,in2,out2,loop2,rel2,q2
  const float* loop1 = (const float*)d_in[12];
  const float* loop2 = (const float*)d_in[13];
  const float* b1    = (const float*)d_in[14];
  const float* b2    = (const float*)d_in[15];
  const int* eidx    = (const int*)d_in[16];
  const int* etyp    = (const int*)d_in[17];
  const int* quals   = (const int*)d_in[18];
  const int* ent_ix  = (const int*)d_in[19];
  const int* rel_ix  = (const int*)d_in[20];
  const int* quals_ix= (const int*)d_in[21];

  // workspace layout
  char* ws = (char*)d_ws;
  size_t off = 0;
  float* node_acc = (float*)(ws + off); off = alignup(off + (size_t)N_ENT * DDIM * 4);
  unsigned short* q_agg = (unsigned short*)(ws + off); off = alignup(off + (size_t)EDIR * DDIM * 2);
  float* nin      = (float*)(ws + off); off = alignup(off + (size_t)EDIR * 4);
  float* nout     = (float*)(ws + off); off = alignup(off + (size_t)EDIR * 4);
  float* dega     = (float*)(ws + off); off = alignup(off + (size_t)N_ENT * 4);
  float* degb     = (float*)(ws + off); off = alignup(off + (size_t)N_ENT * 4);
  float* rel_all  = (float*)(ws + off); off = alignup(off + (size_t)201 * DDIM * 4);
  unsigned short* packw = (unsigned short*)(ws + off);
  off = alignup(off + (size_t)10 * PACK_W_ELEMS * 2);
  (void)ws_size; (void)n_in; (void)in_sizes; (void)out_size;

  // pack all 10 weights to bf16 WMMA-B fragment layout
  for (int i = 0; i < 10; ++i)
    pack_w_kernel<<<KT_TILES * NT_TILES, 32, 0, stream>>>(W[i], packw + (size_t)i * PACK_W_ELEMS);

  // degrees + symmetric norms (layer-invariant)
  const int* src = eidx;              // row0[0:E]
  const int* dst = eidx + 2 * EDIR;   // row1[0:E]
  zero_kernel<<<cdivu(N_ENT, 256), 256, 0, stream>>>(dega, N_ENT);
  zero_kernel<<<cdivu(N_ENT, 256), 256, 0, stream>>>(degb, N_ENT);
  deg_count_kernel<<<cdivu(EDIR, 256), 256, 0, stream>>>(src, dst, dega, degb, EDIR);
  edge_norm_kernel<<<cdivu(EDIR, 256), 256, 0, stream>>>(src, dst, dega, degb, nin, nout, EDIR);

  // d_out layout: sub[B,D] | rel[B,D] | qual_obj[B,6,D] | qual_rel[B,6,D] | x[N,D] | r[200,D]
  float* out = (float*)d_out;
  const size_t OFF_SUB  = 0;
  const size_t OFF_REL  = (size_t)BATCH * DDIM;
  const size_t OFF_QOBJ = OFF_REL + (size_t)BATCH * DDIM;
  const size_t OFF_QREL = OFF_QOBJ + (size_t)BATCH * QPAIR * DDIM;
  const size_t OFF_X    = OFF_QREL + (size_t)BATCH * QPAIR * DDIM;
  const size_t OFF_R    = OFF_X + (size_t)N_ENT * DDIM;
  float* xslot = out + OFF_X;
  float* rslot = out + OFF_R;

  const unsigned short* P = packw;
  // layer 1: x1 -> xslot, r1 -> rslot
  run_layer(stream, ent, rel, loop1, b1,
            P + 0 * PACK_W_ELEMS, P + 1 * PACK_W_ELEMS, P + 2 * PACK_W_ELEMS,
            P + 3 * PACK_W_ELEMS, P + 4 * PACK_W_ELEMS,
            eidx, etyp, quals, nin, nout, q_agg, node_acc, rel_all, xslot, rslot);
  // layer 2: in-place on xslot / rslot
  run_layer(stream, xslot, rslot, loop2, b2,
            P + 5 * PACK_W_ELEMS, P + 6 * PACK_W_ELEMS, P + 7 * PACK_W_ELEMS,
            P + 8 * PACK_W_ELEMS, P + 9 * PACK_W_ELEMS,
            eidx, etyp, quals, nin, nout, q_agg, node_acc, rel_all, xslot, rslot);

  // batch gathers
  gather_kernel<<<cdivu((long)BATCH * DDIM, 256), 256, 0, stream>>>(
      xslot, ent_ix, 1, 0, out + OFF_SUB, BATCH);
  gather_kernel<<<cdivu((long)BATCH * DDIM, 256), 256, 0, stream>>>(
      rslot, rel_ix, 1, 0, out + OFF_REL, BATCH);
  gather_kernel<<<cdivu((long)BATCH * QPAIR * DDIM, 256), 256, 0, stream>>>(
      xslot, quals_ix, 2, 1, out + OFF_QOBJ, BATCH * QPAIR);
  gather_kernel<<<cdivu((long)BATCH * QPAIR * DDIM, 256), 256, 0, stream>>>(
      rslot, quals_ix, 2, 0, out + OFF_QREL, BATCH * QPAIR);
}
